// BiDAF_11922829214355
// MI455X (gfx1250) — matrix operand order repaired and checked by reading, not compiled, not run on hardware
//
#include <hip/hip_runtime.h>
#include <math.h>

// Problem constants (match reference)
static constexpr int kB  = 32;
static constexpr int kCL = 1024;
static constexpr int kQL = 512;
static constexpr int kD  = 256;

// Tiling for main flash kernel
static constexpr int TM = 32;    // c rows per workgroup
static constexpr int TN = 64;    // q rows per chunk
static constexpr int CS = 264;   // LDS stride for c tile (bf16 elems)
static constexpr int QS = 264;   // LDS stride for q chunk (row-major)
static constexpr int TS = 72;    // LDS stride for transposed q chunk
static constexpr int PS = 72;    // LDS stride for probability tile

typedef __attribute__((ext_vector_type(8)))  float          v8f;
typedef __attribute__((ext_vector_type(16))) __bf16         v16bf;
typedef __attribute__((ext_vector_type(16))) unsigned short v16u;

union BF16Frag { v16u u; v16bf b; };

__device__ __forceinline__ unsigned short f2bf(float f) {
    unsigned int u = __float_as_uint(f);
    unsigned int r = u + 0x7FFFu + ((u >> 16) & 1u);   // round-to-nearest-even
    return (unsigned short)(r >> 16);
}

// Async global->LDS 16B copy (CDNA5 GLOBAL_LOAD_ASYNC_TO_LDS_B128, ASYNCcnt).
// saddr form: vdst = LDS byte address, vaddr = 32-bit byte offset, saddr = base.
__device__ __forceinline__ void async_copy_b128(unsigned lds_addr,
                                                unsigned voff,
                                                unsigned long long gbase) {
    asm volatile("global_load_async_to_lds_b128 %0, %1, %2"
                 :: "v"(lds_addr), "v"(voff), "s"(gbase) : "memory");
}

// Wait until at most N async ops remain outstanding (in-order retirement).
template <int N>
__device__ __forceinline__ void wait_async() {
#if defined(__has_builtin)
#if __has_builtin(__builtin_amdgcn_s_wait_asynccnt)
    __builtin_amdgcn_s_wait_asynccnt((short)N);
    return;
#else
    asm volatile("s_wait_asynccnt %0" :: "i"(N) : "memory");
    return;
#endif
#else
    asm volatile("s_wait_asynccnt %0" :: "i"(N) : "memory");
#endif
}

// ---------------------------------------------------------------------------
// k0: sc[b,i] = c[b,i,:].w_c + b_c ;  sq[b,j] = q[b,j,:].w_q + b_q
// One wave32 per row; lane handles 8 consecutive elements of D=256.
// ---------------------------------------------------------------------------
__global__ __launch_bounds__(256) void k0_rowdots(
    const float* __restrict__ c, const float* __restrict__ q,
    const float* __restrict__ w_c, const float* __restrict__ b_c,
    const float* __restrict__ w_q, const float* __restrict__ b_q,
    float* __restrict__ sc, float* __restrict__ sq)
{
    const int tid  = threadIdx.x;
    const int lane = tid & 31;
    const int wave = tid >> 5;
    const int row  = blockIdx.x * 8 + wave;

    const float* src; const float* w; float bias; float* dst; int r;
    if (row < kB * kCL) { src = c; w = w_c; bias = *b_c; dst = sc; r = row; }
    else                { src = q; w = w_q; bias = *b_q; dst = sq; r = row - kB * kCL; }

    const float4* p  = (const float4*)(src + (size_t)r * kD + lane * 8);
    const float4* wp = (const float4*)(w + lane * 8);
    float4 a0 = p[0], a1 = p[1], w0 = wp[0], w1 = wp[1];
    float s = a0.x*w0.x + a0.y*w0.y + a0.z*w0.z + a0.w*w0.w
            + a1.x*w1.x + a1.y*w1.y + a1.z*w1.z + a1.w*w1.w;
#pragma unroll
    for (int m = 1; m < 32; m <<= 1) s += __shfl_xor(s, m, 32);
    if (lane == 0) dst[r] = s + bias;
}

// ---------------------------------------------------------------------------
// k0q: one-time q -> bf16, row-major (qbf[b][j][d]) and transposed
// (qTg[b][d][j]). 64x64 tiles via LDS so both global writes are coalesced.
// Grid: (QL/64, D/64, B), 256 threads.
// ---------------------------------------------------------------------------
__global__ __launch_bounds__(256) void k0_qconv(
    const float* __restrict__ q, unsigned short* __restrict__ qbf,
    unsigned short* __restrict__ qTg)
{
    __shared__ __align__(16) unsigned short tile[64][72];
    const int tid = threadIdx.x;
    const int b  = blockIdx.z;
    const int j0 = blockIdx.x * 64;
    const int d0 = blockIdx.y * 64;
    const float* qb = q + ((size_t)b * kQL + j0) * kD + d0;

    for (int idx = tid; idx < 64 * 16; idx += 256) {   // 64 rows x 16 quads
        const int rr = idx >> 4, c4 = (idx & 15) * 4;
        float4 v = *(const float4*)(qb + (size_t)rr * kD + c4);
        ushort4 o;
        o.x = f2bf(v.x); o.y = f2bf(v.y); o.z = f2bf(v.z); o.w = f2bf(v.w);
        *(ushort4*)&qbf[((size_t)b * kQL + j0 + rr) * kD + d0 + c4] = o;
        *(ushort4*)&tile[rr][c4] = o;
    }
    __syncthreads();
    for (int idx = tid; idx < 64 * 16; idx += 256) {   // 64 d-rows x 16 quads
        const int dd = idx >> 4, j4 = (idx & 15) * 4;
        ushort4 o;
        o.x = tile[j4 + 0][dd]; o.y = tile[j4 + 1][dd];
        o.z = tile[j4 + 2][dd]; o.w = tile[j4 + 3][dd];
        *(ushort4*)&qTg[((size_t)b * kD + d0 + dd) * kQL + j0 + j4] = o;
    }
}

// ---------------------------------------------------------------------------
// k1: flash-attention style main kernel with bf16 WMMA + async LDS staging.
// Grid (CL/TM, B), 256 threads = 8 waves.
// Computes c2q = softmax_j(sq_j + (c*w_cq).q_j) @ q  and rowmax for bvec.
// Writes out[...,0:D]=c, [D:2D]=c2q, [2D:3D]=c*c2q.
// ---------------------------------------------------------------------------
__global__ __launch_bounds__(256) void k1_flash(
    const float* __restrict__ c,
    const unsigned short* __restrict__ qbf,   // q bf16 row-major
    const unsigned short* __restrict__ qTg,   // q bf16 transposed [b][d][j]
    const float* __restrict__ w_cq,
    const float* __restrict__ sc_ws, const float* __restrict__ sq_ws,
    float* __restrict__ out, float* __restrict__ rowmax_ws)
{
    __shared__ __align__(16) unsigned short cs_lds[TM][CS]; // (c*w_cq) bf16
    __shared__ __align__(16) unsigned short q_lds[TN][QS];  // q chunk, row-major
    __shared__ __align__(16) unsigned short qT_lds[kD][TS]; // q chunk, transposed
    __shared__ __align__(16) unsigned short p_lds[TM][PS];  // probs bf16
    __shared__ float m_lds[TM], l_lds[TM], scl_lds[TM], part[4][TM];

    const int tid  = threadIdx.x;
    const int lane = tid & 31;
    const int wave = tid >> 5;
    const int half = lane >> 4;     // 0: lanes 0-15, 1: lanes 16-31
    const int n    = lane & 15;
    const int b     = blockIdx.y;
    const int ibase = blockIdx.x * TM;
    const int r  = wave >> 2;       // row-subtile for GEMM1 (0..1)
    const int jt = wave & 3;        // col-subtile for GEMM1 (0..3)

    // ---- load c tile (float4), scale by w_cq, convert to bf16 ----
    const float* cb = c + ((size_t)b * kCL + ibase) * kD;
    for (int idx = tid; idx < TM * kD / 4; idx += 256) {
        const int rr = idx >> 6;            // kD/4 = 64 quads per row
        const int c4 = (idx & 63) * 4;
        float4 v = *(const float4*)(cb + (size_t)rr * kD + c4);
        float4 w = *(const float4*)(w_cq + c4);
        ushort4 o;
        o.x = f2bf(v.x * w.x); o.y = f2bf(v.y * w.y);
        o.z = f2bf(v.z * w.z); o.w = f2bf(v.w * w.w);
        *(ushort4*)&cs_lds[rr][c4] = o;
    }
    if (tid < TM) { m_lds[tid] = -3.4e38f; l_lds[tid] = 0.f; }

    v8f zero8 = {0.f,0.f,0.f,0.f,0.f,0.f,0.f,0.f};
    v8f acc[2][2];                  // O accumulators: [rowhalf][ctl]
    acc[0][0] = zero8; acc[0][1] = zero8; acc[1][0] = zero8; acc[1][1] = zero8;

    const unsigned long long qbf_base = (unsigned long long)(qbf + (size_t)b * kQL * kD);
    const unsigned long long qTg_base = (unsigned long long)(qTg + (size_t)b * kD * kQL);
    const float* sqb = sq_ws + b * kQL;

    for (int jc = 0; jc < kQL / TN; ++jc) {
        const int jbase = jc * TN;
        __syncthreads();   // prev chunk's LDS reads done before overwrite

        // ---- async-stage q chunk: row-major (TN x kD), 8 ops/thread ----
        for (int t = tid; t < TN * kD / 8; t += 256) {
            const int rr = t >> 5;          // kD/8 = 32 segs per row
            const int sg = (t & 31) * 8;
            const unsigned voff = (unsigned)(((jbase + rr) * kD + sg) * 2);
            const unsigned la   = (unsigned)(uintptr_t)&q_lds[rr][sg];
            async_copy_b128(la, voff, qbf_base);
        }
        // ---- async-stage q chunk transposed (kD x TN), 8 ops/thread ----
        for (int t = tid; t < kD * TN / 8; t += 256) {
            const int dd = t >> 3;          // TN/8 = 8 segs per d-row
            const int sg = (t & 7) * 8;
            const unsigned voff = (unsigned)((dd * kQL + jbase + sg) * 2);
            const unsigned la   = (unsigned)(uintptr_t)&qT_lds[dd][sg];
            async_copy_b128(la, voff, qTg_base);
        }
        // Async ops retire in order: <=8 outstanding means the row-major group
        // has landed; the transposed group may still be in flight (only needed
        // by GEMM2, waited on below). Overlaps DMA with GEMM1 + softmax.
        wait_async<8>();
        __syncthreads();

        // ---- GEMM1: S(16x16) = (c*w_cq) @ q^T over K=256 ----
        v8f s8 = zero8;
        const unsigned short* arow = &cs_lds[r * 16 + n][0];
        const unsigned short* brow = &q_lds[jt * 16 + n][0];
#pragma unroll
        for (int kb = 0; kb < kD / 32; ++kb) {
            const int kbase = kb * 32;
            BF16Frag A, Bq;
#pragma unroll
            for (int e = 0; e < 8; ++e) {
                A.u[e]     = arow[kbase + 16 * half + e];       // K = e + 16*half
                A.u[e + 8] = arow[kbase + 16 * half + 8 + e];   // K = (e+8) + 16*half
            }
#pragma unroll
            for (int e = 0; e < 16; ++e)
                Bq.u[e] = brow[kbase + 16 * half + e];          // K = e + 16*half
            s8 = __builtin_amdgcn_wmma_f32_16x16x32_bf16(
                     false, A.b, false, Bq.b, (short)0, s8, false, false);
        }
        // add sq_j (column bias; sc_i/b_cq drop out of softmax over j)
        const float sqv = sqb[jbase + jt * 16 + n];
#pragma unroll
        for (int v = 0; v < 8; ++v) s8[v] += sqv;

        // ---- per-row chunk max (reduce across 16 lanes per half) ----
#pragma unroll
        for (int v = 0; v < 8; ++v) {
            float x = s8[v];
            x = fmaxf(x, __shfl_xor(x, 1, 32));
            x = fmaxf(x, __shfl_xor(x, 2, 32));
            x = fmaxf(x, __shfl_xor(x, 4, 32));
            x = fmaxf(x, __shfl_xor(x, 8, 32));
            if (n == 0) part[jt][r * 16 + v + 8 * half] = x;
        }
        __syncthreads();
        if (tid < TM) {
            float mo = m_lds[tid];
            float cm = fmaxf(fmaxf(part[0][tid], part[1][tid]),
                             fmaxf(part[2][tid], part[3][tid]));
            float mn = fmaxf(mo, cm);
            m_lds[tid]   = mn;
            scl_lds[tid] = __expf(mo - mn);   // 0 on first chunk
        }
        __syncthreads();

        // ---- P = exp(S - m) ; row sums ; bf16 store ----
#pragma unroll
        for (int v = 0; v < 8; ++v) {
            const int row = r * 16 + v + 8 * half;
            float p = __expf(s8[v] - m_lds[row]);
            p_lds[row][jt * 16 + n] = f2bf(p);
            float x = p;
            x += __shfl_xor(x, 1, 32);
            x += __shfl_xor(x, 2, 32);
            x += __shfl_xor(x, 4, 32);
            x += __shfl_xor(x, 8, 32);
            if (n == 0) part[jt][row] = x;
        }
        wait_async<0>();   // transposed q chunk now fully in LDS
        __syncthreads();
        if (tid < TM)
            l_lds[tid] = l_lds[tid] * scl_lds[tid] +
                         part[0][tid] + part[1][tid] + part[2][tid] + part[3][tid];

        // ---- rescale O accumulators by exp(m_old - m_new) ----
#pragma unroll
        for (int rp = 0; rp < 2; ++rp) {
#pragma unroll
            for (int v = 0; v < 8; ++v) {
                const float sclv = scl_lds[rp * 16 + v + 8 * half];
                acc[rp][0][v] *= sclv;
                acc[rp][1][v] *= sclv;
            }
        }

        // ---- GEMM2: O += P(32xTN) @ Qchunk(TNxD), B from transposed LDS ----
#pragma unroll
        for (int rp = 0; rp < 2; ++rp) {
            const unsigned short* prow = &p_lds[rp * 16 + n][0];
#pragma unroll
            for (int kb = 0; kb < TN / 32; ++kb) {
                const int kbase = kb * 32;
                BF16Frag A;
#pragma unroll
                for (int e = 0; e < 8; ++e) {
                    A.u[e]     = prow[kbase + 16 * half + e];
                    A.u[e + 8] = prow[kbase + 16 * half + 8 + e];
                }
#pragma unroll
                for (int ctl = 0; ctl < 2; ++ctl) {
                    const int dcol = (wave * 2 + ctl) * 16 + n;
                    const unsigned short* trow = &qT_lds[dcol][0];
                    BF16Frag Bq;
#pragma unroll
                    for (int e = 0; e < 16; ++e)
                        Bq.u[e] = trow[kbase + 16 * half + e];   // K = e + 16*half
                    acc[rp][ctl] = __builtin_amdgcn_wmma_f32_16x16x32_bf16(
                        false, A.b, false, Bq.b, (short)0, acc[rp][ctl], false, false);
                }
            }
        }
    }

    __syncthreads();
    if (tid < TM) {
        scl_lds[tid] = 1.0f / l_lds[tid];   // reuse as 1/l
        // rowmax for bvec: max_j(sq_j + cq_ij) + sc_i  (b_q/b_cq shift out)
        rowmax_ws[(size_t)b * kCL + ibase + tid] =
            m_lds[tid] + sc_ws[(size_t)b * kCL + ibase + tid];
    }
    __syncthreads();

    // ---- epilogue: write c, c2q, c*c2q (non-temporal: out never re-read) ----
#pragma unroll
    for (int rp = 0; rp < 2; ++rp) {
#pragma unroll
        for (int ctl = 0; ctl < 2; ++ctl) {
#pragma unroll
            for (int v = 0; v < 8; ++v) {
                const int row = rp * 16 + v + 8 * half;
                const int i   = ibase + row;
                const int d   = (wave * 2 + ctl) * 16 + n;
                const float o  = acc[rp][ctl][v] * scl_lds[row];
                const float cv = c[((size_t)b * kCL + i) * kD + d];
                const size_t ob = ((size_t)b * kCL + i) * (size_t)(4 * kD);
                __builtin_nontemporal_store(cv,      &out[ob + d]);
                __builtin_nontemporal_store(o,       &out[ob + kD + d]);
                __builtin_nontemporal_store(cv * o,  &out[ob + 2 * kD + d]);
            }
        }
    }
}

// ---------------------------------------------------------------------------
// k2: per batch: bvec = softmax_i(rowmax_i), q2c[d] = sum_i bvec_i * c[b,i,d]
// 256 threads: one thread per d.
// ---------------------------------------------------------------------------
__global__ __launch_bounds__(256) void k2_q2c(
    const float* __restrict__ rowmax_ws, const float* __restrict__ c,
    float* __restrict__ q2c)
{
    __shared__ float bw[kCL];
    __shared__ float wred[8];
    const int b = blockIdx.x, tid = threadIdx.x;
    const int lane = tid & 31, wave = tid >> 5;
    const float* rmb = rowmax_ws + (size_t)b * kCL;

    float v = -3.4e38f;
    for (int i = tid; i < kCL; i += 256) v = fmaxf(v, rmb[i]);
#pragma unroll
    for (int m = 1; m < 32; m <<= 1) v = fmaxf(v, __shfl_xor(v, m, 32));
    if (lane == 0) wred[wave] = v;
    __syncthreads();
    float bm = wred[0];
#pragma unroll
    for (int w = 1; w < 8; ++w) bm = fmaxf(bm, wred[w]);
    __syncthreads();

    float s = 0.f;
    for (int i = tid; i < kCL; i += 256) {
        float e = __expf(rmb[i] - bm);
        bw[i] = e;
        s += e;
    }
#pragma unroll
    for (int m = 1; m < 32; m <<= 1) s += __shfl_xor(s, m, 32);
    if (lane == 0) wred[wave] = s;
    __syncthreads();
    float tot = 0.f;
#pragma unroll
    for (int w = 0; w < 8; ++w) tot += wred[w];
    const float inv = 1.0f / tot;

    const int d = tid;                    // D == 256 == blockDim
    const float* cbb = c + (size_t)b * kCL * kD;
    float accv = 0.f;
    for (int i = 0; i < kCL; ++i) accv += bw[i] * cbb[(size_t)i * kD + d];
    q2c[b * kD + d] = accv * inv;
}

// ---------------------------------------------------------------------------
// k3: out[b,i,3D+d] = c[b,i,d] * q2c[b,d]
// ---------------------------------------------------------------------------
__global__ __launch_bounds__(256) void k3_q2catt(
    const float* __restrict__ c, const float* __restrict__ q2c,
    float* __restrict__ out)
{
    const size_t idx = (size_t)blockIdx.x * 256 + threadIdx.x;  // b*CL*D total
    const int d = (int)(idx & 255);
    const size_t row = idx >> 8;            // b*CL + i
    const int b = (int)(row >> 10);         // CL = 1024
    const float val = c[idx] * q2c[b * kD + d];
    __builtin_nontemporal_store(val, &out[row * (size_t)(4 * kD) + 3 * kD + d]);
}

// ---------------------------------------------------------------------------
extern "C" void kernel_launch(void* const* d_in, const int* in_sizes, int n_in,
                              void* d_out, int out_size, void* d_ws, size_t ws_size,
                              hipStream_t stream)
{
    const float* c    = (const float*)d_in[0];
    const float* q    = (const float*)d_in[1];
    const float* w_c  = (const float*)d_in[2];
    const float* b_c  = (const float*)d_in[3];
    const float* w_q  = (const float*)d_in[4];
    const float* b_q  = (const float*)d_in[5];
    const float* w_cq = (const float*)d_in[6];
    // d_in[7] = b_cq: provably has no effect on the output (softmax shift
    // invariance over both axes) -> unused.
    float* out = (float*)d_out;

    float* ws   = (float*)d_ws;
    float* sc   = ws;                              // B*CL
    float* sq   = sc + kB * kCL;                   // B*QL
    float* rmax = sq + kB * kQL;                   // B*CL
    float* q2c  = rmax + kB * kCL;                 // B*D
    unsigned short* qbf = (unsigned short*)(q2c + kB * kD);        // B*QL*D bf16
    unsigned short* qTg = qbf + (size_t)kB * kQL * kD;             // B*D*QL bf16

    k0_rowdots<<<(kB * kCL + kB * kQL) / 8, 256, 0, stream>>>(
        c, q, w_c, b_c, w_q, b_q, sc, sq);
    k0_qconv<<<dim3(kQL / 64, kD / 64, kB), 256, 0, stream>>>(q, qbf, qTg);
    k1_flash<<<dim3(kCL / TM, kB), 256, 0, stream>>>(
        c, qbf, qTg, w_cq, sc, sq, out, rmax);
    k2_q2c<<<kB, 256, 0, stream>>>(rmax, c, q2c);
    k3_q2catt<<<(kB * kCL * kD) / 256, 256, 0, stream>>>(c, q2c, out);
}
